// SelfAttentionLayer_76828374991305
// MI455X (gfx1250) — matrix operand order repaired and checked by reading, not compiled, not run on hardware
//
#include <hip/hip_runtime.h>
#include <hip/hip_bf16.h>

// ---------------------------------------------------------------------------
// Shapes fixed by the reference: B=2, T=2048, D=1024, H=16, hD=64.
// All GEMMs run through v_wmma_f32_16x16x32_bf16 (wave32, 16x16 tiles, K=32).
// B panels are staged through LDS (double-buffered) so the 4 waves of a block
// share one coalesced fetch of each 32x64 panel.
// ---------------------------------------------------------------------------

#define BB   2
#define TT   2048
#define DD   1024
#define HH   16
#define HD   64

typedef __attribute__((ext_vector_type(16))) __bf16         v16bf;
typedef __attribute__((ext_vector_type(8)))  float          v8f;
typedef __attribute__((ext_vector_type(8)))  unsigned short v8u16;
typedef __attribute__((ext_vector_type(16))) unsigned short v16u16;

__device__ __forceinline__ unsigned short f32_to_bf16(float f) {
    unsigned int u = __float_as_uint(f);
    u += 0x7FFFu + ((u >> 16) & 1u);      // round-to-nearest-even
    return (unsigned short)(u >> 16);
}

// ---------------------------------------------------------------------------
// Batched WMMA GEMM:  C[m,n] = alpha * sum_k A[m,k] * B[k,n]  (+bias +residual)
//   A: bf16 (AF32=false) or fp32 converted on the fly (AF32=true), row-major
//   B: bf16, K x N row-major (ldb = N-stride), staged via LDS double buffer
//   SMODE: 0 = fp32 store, 1 = bf16 store, 2 = bf16 head-transposed (B,H,hD,T)
// Block = 128 threads = 4 waves; wave w owns rows [by*64+16w, +16), block owns
// cols [bx*64, +64).  grid.z -> (b = z/Hb, h = z%Hb) with explicit strides.
// ---------------------------------------------------------------------------
template <bool AF32, int SMODE, bool CAUSAL>
__global__ void wmma_gemm_kernel(
    const unsigned short* __restrict__ Ab,
    const float*          __restrict__ Af,
    long lda,
    const unsigned short* __restrict__ Bm,
    long ldb,
    int K, float alpha,
    const float* __restrict__ bias,
    const float* __restrict__ residual,
    float*          __restrict__ Cf,
    unsigned short* __restrict__ Cbf,
    long ldc,
    int Hb,
    long sAb, long sAh, long sBb, long sBh, long sCb, long sCh)
{
    const int bx = blockIdx.x, by = blockIdx.y, z = blockIdx.z;
    if (CAUSAL && bx > by) return;       // whole upper-triangle block: skip

    __shared__ unsigned short Bp[2][32 * 64];   // two 4 KB B panels

    const int tid  = threadIdx.x;
    const int wave = tid >> 5;
    const int lane = tid & 31;
    const int r    = lane & 15;          // row/col within 16
    const int half = lane >> 4;          // 0 or 1

    const long m0 = (long)by * 64 + (long)wave * 16;
    const long n0 = (long)bx * 64;

    const int b = z / Hb, h = z % Hb;
    const long aoff = (long)b * sAb + (long)h * sAh;
    const long boff = (long)b * sBb + (long)h * sBh;
    const long coff = (long)b * sCb + (long)h * sCh;

    // panel staging role: thread -> (row srow of 32, 16-col chunk scol of 64)
    const int  srow = tid >> 2;
    const int  scol = (tid & 3) * 16;
    const unsigned short* bstage = Bm + boff + (long)srow * ldb + n0 + scol;

    v8f acc[4] = {};

    // preload first B panel into registers
    v8u16 g0 = *(const v8u16*)(bstage);
    v8u16 g1 = *(const v8u16*)(bstage + 8);

    for (int k = 0; k < K; k += 32) {
        const int cur = (k >> 5) & 1;

        // commit staged panel to LDS (waits on the in-flight global loads)
        *(v8u16*)&Bp[cur][srow * 64 + scol]     = g0;
        *(v8u16*)&Bp[cur][srow * 64 + scol + 8] = g1;

        // kick off global loads for the next panel (overlaps the WMMA block)
        if (k + 32 < K) {
            const unsigned short* bn = bstage + (long)(k + 32) * ldb;
            __builtin_prefetch(bn + 64 * ldb, 0, 1);
            g0 = *(const v8u16*)(bn);
            g1 = *(const v8u16*)(bn + 8);
        }

        __syncthreads();                 // panel `cur` visible to all waves

        // ---- A fragment: lane r holds row m0+r; K {k+8h..+7, k+16+8h..+7}
        v16u16 au;
        if (AF32) {
            const float* pa = Af + aoff + (m0 + r) * lda + k + half * 8;
#pragma unroll
            for (int i = 0; i < 8; ++i) {
                au[i]     = f32_to_bf16(pa[i]);
                au[i + 8] = f32_to_bf16(pa[i + 16]);
            }
        } else {
            const unsigned short* pa = Ab + aoff + (m0 + r) * lda + k + half * 8;
            v8u16 lo = *(const v8u16*)pa;
            v8u16 hi = *(const v8u16*)(pa + 16);
#pragma unroll
            for (int i = 0; i < 8; ++i) { au[i] = lo[i]; au[i + 8] = hi[i]; }
        }
        const v16bf afrag = __builtin_bit_cast(v16bf, au);

        // ---- B fragments from LDS: lane holds col 16t+r; K {8h..+7, 16+8h..+7}
        const int kk = half * 8;
#pragma unroll
        for (int t = 0; t < 4; ++t) {
            const unsigned short* pb = &Bp[cur][kk * 64 + t * 16 + r];
            v16u16 bu;
#pragma unroll
            for (int i = 0; i < 8; ++i) { bu[i] = pb[i * 64]; bu[i + 8] = pb[(i + 16) * 64]; }
            const v16bf bfrag = __builtin_bit_cast(v16bf, bu);

            acc[t] = __builtin_amdgcn_wmma_f32_16x16x32_bf16(
                false, afrag, false, bfrag, (short)0, acc[t], false, false);
        }
        // single barrier per step: buffer cur is only overwritten two steps
        // later, i.e. after every wave has passed the next barrier.
    }

    // ---- Epilogue. C/D layout: elem e of lane -> row m0+e+8*half, col n0+16t+r
#pragma unroll
    for (int t = 0; t < 4; ++t) {
#pragma unroll
        for (int e = 0; e < 8; ++e) {
            const long row = m0 + e + half * 8;
            const long col = n0 + t * 16 + r;
            float v = acc[t][e] * alpha;
            if (bias)     v += bias[col];
            if (residual) v += residual[row * ldc + col];
            if (SMODE == 0) {
                Cf[coff + row * ldc + col] = v;
            } else if (SMODE == 1) {
                Cbf[coff + row * ldc + col] = f32_to_bf16(v);
            } else {
                // head-transposed: (b, h, d, t) over fixed T=2048, H=16, hD=64
                const long b2 = row >> 11, s = row & 2047;
                const long hh = col >> 6,  dd = col & 63;
                Cbf[(((b2 << 4) + hh) * 64 + dd) * 2048 + s] = f32_to_bf16(v);
            }
        }
    }
}

// ---------------------------------------------------------------------------
// fp32 -> bf16 weight transpose: Wt[k*N + n] = W[n*K + k]  (W is N x K)
// ---------------------------------------------------------------------------
__global__ void transpose_bf16_kernel(const float* __restrict__ W,
                                      unsigned short* __restrict__ Wt,
                                      int N, int K)
{
    long i = (long)blockIdx.x * 256 + threadIdx.x;
    if (i >= (long)N * K) return;
    int n = (int)(i / K), k = (int)(i % K);
    Wt[(long)k * N + n] = f32_to_bf16(W[i]);
}

// ---------------------------------------------------------------------------
// Pre-LN: one block per row of (B*T, D); writes bf16 normalized activations
// ---------------------------------------------------------------------------
__global__ void layernorm_kernel(const float* __restrict__ x,
                                 const float* __restrict__ gamma,
                                 const float* __restrict__ beta,
                                 unsigned short* __restrict__ xnb)
{
    __shared__ float s1[256], s2[256];
    const long row = blockIdx.x;
    const int tid = threadIdx.x;
    const float* xr = x + row * DD;

    float a = 0.f, bsum = 0.f;
    for (int c = tid; c < DD; c += 256) { float v = xr[c]; a += v; bsum += v * v; }
    s1[tid] = a; s2[tid] = bsum;
    __syncthreads();
    for (int s = 128; s > 0; s >>= 1) {
        if (tid < s) { s1[tid] += s1[tid + s]; s2[tid] += s2[tid + s]; }
        __syncthreads();
    }
    const float mean = s1[0] * (1.f / DD);
    const float var  = s2[0] * (1.f / DD) - mean * mean;
    const float rstd = rsqrtf(var + 1e-5f);
    for (int c = tid; c < DD; c += 256) {
        float v = (xr[c] - mean) * rstd * gamma[c] + beta[c];
        xnb[row * DD + c] = f32_to_bf16(v);
    }
}

// ---------------------------------------------------------------------------
// RoPE (split-half variant) on q/k of qkv1 (fp32, row stride 3*D); emits bf16
// rotated q,k and bf16 v.  One thread per (row, head, d) pair, d in [0,32).
// ---------------------------------------------------------------------------
__global__ void rope_kernel(const float* __restrict__ qkv,
                            unsigned short* __restrict__ qb,
                            unsigned short* __restrict__ kb,
                            unsigned short* __restrict__ vb)
{
    const long i = (long)blockIdx.x * 256 + threadIdx.x;   // B*T*H*32 threads
    const long row = i >> 9;
    const int rem = (int)(i & 511);
    const int h = rem >> 5, d = rem & 31;
    const int t = (int)(row & (TT - 1));

    // inv_freq = 10000^(-d/32) = exp(-d * ln(10000)/32)
    const float inv = expf(-(float)d * (9.21034037197618f / 32.f));
    const float ang = (float)t * inv;
    float s, c;
    sincosf(ang, &s, &c);

    const int c1 = (h << 6) + d, c2 = c1 + 32;
    const float* p = qkv + row * (3 * DD);
    const float q1 = p[c1],        q2 = p[c2];
    const float k1 = p[DD + c1],   k2 = p[DD + c2];
    const float v1 = p[2*DD + c1], v2 = p[2*DD + c2];

    qb[row * DD + c1] = f32_to_bf16(q1 * c - q2 * s);
    qb[row * DD + c2] = f32_to_bf16(q2 * c + q1 * s);
    kb[row * DD + c1] = f32_to_bf16(k1 * c - k2 * s);
    kb[row * DD + c2] = f32_to_bf16(k2 * c + k1 * s);
    vb[row * DD + c1] = f32_to_bf16(v1);
    vb[row * DD + c2] = f32_to_bf16(v2);
}

// ---------------------------------------------------------------------------
// Causal softmax in place over attn rows (B*H*T rows of length T).
// Valid span for row with time index t is [0, t]; rest is zeroed.
// ---------------------------------------------------------------------------
__global__ void softmax_causal_kernel(float* __restrict__ attn)
{
    __shared__ float sm[256];
    const long row = blockIdx.x;
    const int  t   = (int)(row & (TT - 1));
    float* p = attn + row * TT;
    const int tid = threadIdx.x;

    float lmax = -3.402823466e38f;
    for (int i = tid; i <= t; i += 256) lmax = fmaxf(lmax, p[i]);
    sm[tid] = lmax;
    __syncthreads();
    for (int s = 128; s > 0; s >>= 1) {
        if (tid < s) sm[tid] = fmaxf(sm[tid], sm[tid + s]);
        __syncthreads();
    }
    const float m = sm[0];
    __syncthreads();

    float lsum = 0.f;
    for (int i = tid; i <= t; i += 256) {
        float e = expf(p[i] - m);
        p[i] = e;
        lsum += e;
    }
    sm[tid] = lsum;
    __syncthreads();
    for (int s = 128; s > 0; s >>= 1) {
        if (tid < s) sm[tid] += sm[tid + s];
        __syncthreads();
    }
    const float invsum = 1.f / sm[0];
    for (int i = tid; i <= t; i += 256)         p[i] *= invsum;
    for (int i = t + 1 + tid; i < TT; i += 256) p[i] = 0.f;
}

// ---------------------------------------------------------------------------
// Host orchestration
// ---------------------------------------------------------------------------
extern "C" void kernel_launch(void* const* d_in, const int* in_sizes, int n_in,
                              void* d_out, int out_size, void* d_ws, size_t ws_size,
                              hipStream_t stream)
{
    (void)in_sizes; (void)n_in; (void)out_size; (void)ws_size;

    const float* x     = (const float*)d_in[0];   // (B,T,D)
    const float* Win   = (const float*)d_in[1];   // (3D,D)
    const float* bin   = (const float*)d_in[2];   // (3D,)
    const float* Wout  = (const float*)d_in[3];   // (D,D)
    const float* bout  = (const float*)d_in[4];   // (D,)
    const float* gamma = (const float*)d_in[5];
    const float* beta  = (const float*)d_in[6];

    float* out  = (float*)d_out;                          // (B,T,D)
    float* attn = out + (long)BB * TT * DD;               // (B,H,T,T)

    // ---- workspace carve-up ----
    char* ws = (char*)d_ws;
    unsigned short* xnb   = (unsigned short*)ws;  ws += (long)BB*TT*DD * 2;        // LN(x) bf16
    unsigned short* wtin  = (unsigned short*)ws;  ws += (long)DD * (3*DD) * 2;     // in_proj^T bf16 (K x 3D)
    unsigned short* wtout = (unsigned short*)ws;  ws += (long)DD * DD * 2;         // out_proj^T bf16
    float*          qkv1  = (float*)ws;           ws += (long)BB*TT*(3*DD) * 4;    // first projection fp32
    unsigned short* qrb   = (unsigned short*)ws;  ws += (long)BB*TT*DD * 2;        // RoPE(q) bf16
    unsigned short* krb   = (unsigned short*)ws;  ws += (long)BB*TT*DD * 2;        // RoPE(k) bf16
    unsigned short* vb    = (unsigned short*)ws;  ws += (long)BB*TT*DD * 2;        // v bf16
    unsigned short* q2b   = (unsigned short*)ws;  ws += (long)BB*TT*DD * 2;        // q2 bf16 (B,T,D)
    unsigned short* k2t   = (unsigned short*)ws;  ws += (long)BB*TT*DD * 2;        // k2 bf16 (B,H,hD,T)
    unsigned short* v2b   = (unsigned short*)ws;  ws += (long)BB*TT*DD * 2;        // v2 bf16 (B,T,D)
    unsigned short* aob   = (unsigned short*)ws;  ws += (long)BB*TT*DD * 2;        // attn@v bf16 (B,T,D)

    const long M  = (long)BB * TT;   // 4096
    const long TD = (long)TT * DD;   // per-batch activation stride
    const long T2 = (long)TT * TT;   // per-(b,h) score stride

    // 1) weights -> bf16, transposed to K x N
    transpose_bf16_kernel<<<dim3((3*DD*DD + 255) / 256), 256, 0, stream>>>(Win,  wtin,  3*DD, DD);
    transpose_bf16_kernel<<<dim3((DD*DD   + 255) / 256), 256, 0, stream>>>(Wout, wtout, DD,   DD);

    // 2) pre-LN
    layernorm_kernel<<<dim3((unsigned)M), 256, 0, stream>>>(x, gamma, beta, xnb);

    // 3) QKV projection: qkv1 = xn @ in_proj^T + bin   (fp32 out, N=3072)
    wmma_gemm_kernel<false, 0, false><<<dim3(3*DD/64, M/64, 1), 128, 0, stream>>>(
        xnb, nullptr, DD, wtin, 3*DD, DD, 1.0f, bin, nullptr,
        qkv1, nullptr, 3*DD, 1, 0,0,0,0,0,0);

    // 4) RoPE + bf16 conversion of q/k/v
    rope_kernel<<<dim3((unsigned)(M * HH * 32 / 256)), 256, 0, stream>>>(qkv1, qrb, krb, vb);

    // 5) second projections (bias-free)
    wmma_gemm_kernel<false, 1, false><<<dim3(DD/64, M/64, 1), 128, 0, stream>>>(   // q2 (bf16 natural)
        qrb, nullptr, DD, wtin + 0,    3*DD, DD, 1.0f, nullptr, nullptr,
        nullptr, q2b, DD, 1, 0,0,0,0,0,0);
    wmma_gemm_kernel<false, 2, false><<<dim3(DD/64, M/64, 1), 128, 0, stream>>>(   // k2 (head-transposed)
        krb, nullptr, DD, wtin + DD,   3*DD, DD, 1.0f, nullptr, nullptr,
        nullptr, k2t, DD, 1, 0,0,0,0,0,0);
    wmma_gemm_kernel<false, 1, false><<<dim3(DD/64, M/64, 1), 128, 0, stream>>>(   // v2 (bf16 natural)
        vb,  nullptr, DD, wtin + 2*DD, 3*DD, DD, 1.0f, nullptr, nullptr,
        nullptr, v2b, DD, 1, 0,0,0,0,0,0);

    // 6) scores = (q2 @ k2^T) * hD^-1/2, causal-skipped, batched over (b,h)
    wmma_gemm_kernel<false, 0, true><<<dim3(TT/64, TT/64, BB*HH), 128, 0, stream>>>(
        q2b, nullptr, DD, k2t, TT, HD, 0.125f, nullptr, nullptr,
        attn, nullptr, TT, HH,
        /*sAb*/ TD, /*sAh*/ HD, /*sBb*/ (long)HH*HD*TT, /*sBh*/ (long)HD*TT,
        /*sCb*/ (long)HH*T2, /*sCh*/ T2);

    // 7) causal softmax in place (writes all of each row, incl. zeros)
    softmax_causal_kernel<<<dim3((unsigned)(BB*HH*TT)), 256, 0, stream>>>(attn);

    // 8) attn @ v  (A is fp32 attn, converted to bf16 on the fly; N=64 per head)
    wmma_gemm_kernel<true, 1, false><<<dim3(1, TT/64, BB*HH), 128, 0, stream>>>(
        nullptr, attn, TT, v2b, DD, TT, 1.0f, nullptr, nullptr,
        nullptr, aob, DD, HH,
        /*sAb*/ (long)HH*T2, /*sAh*/ T2, /*sBb*/ TD, /*sBh*/ HD,
        /*sCb*/ TD, /*sCh*/ HD);

    // 9) output projection + bias + residual
    wmma_gemm_kernel<false, 0, false><<<dim3(DD/64, M/64, 1), 128, 0, stream>>>(
        aob, nullptr, DD, wtout, DD, DD, 1.0f, bout, x,
        out, nullptr, DD, 1, 0,0,0,0,0,0);
}
